// SelfAttention_55748675502567
// MI455X (gfx1250) — matrix operand order repaired and checked by reading, not compiled, not run on hardware
//
#include <hip/hip_runtime.h>
#include <hip/hip_bf16.h>

typedef __attribute__((ext_vector_type(16))) _Float16 v16h;
typedef __attribute__((ext_vector_type(8)))  _Float16 v8h;
typedef __attribute__((ext_vector_type(8)))  float    v8f;
typedef __attribute__((ext_vector_type(4)))  float    v4f;

#define B_N 16
#define S_N 2048
#define D_N 128
#define KT  32            // keys per tile
#define NT  (S_N / KT)    // 64 key tiles
#define WAVES 4           // waves per block, one 16-query tile each
#define MASK_SENTINEL (-100)

__global__ __launch_bounds__(128, 1) void fa_fwd_wmma(
    const float* __restrict__ Vg,   // value  [B,S,D]
    const float* __restrict__ Kg,   // key    [B,S,D]
    const float* __restrict__ Qg,   // query  [B,S,D]
    const int*   __restrict__ Mg,   // mask   [B,S,S]
    float*       __restrict__ Og)   // out    [B,S,D]
{
    __shared__ _Float16 Kl[KT][D_N];         // 8 KB : K tile, [key][d]
    __shared__ _Float16 Vt[D_N][KT];         // 8 KB : V tile transposed, [d][key]
    __shared__ _Float16 Pl[WAVES][16][KT];   // 4 KB : per-wave P staging

    const int tid  = threadIdx.x;
    const int wave = tid >> 5;
    const int lane = tid & 31;
    const int lh   = lane & 15;   // column within 16-wide fragment
    const int hi   = lane >> 4;   // which half of the wave

    const int bb = blockIdx.x >> 5;        // batch (32 q-blocks per batch)
    const int qb = blockIdx.x & 31;
    const int q0 = qb * (WAVES * 16) + wave * 16;

    const float scale = 0.08838834764831843f;  // 1/sqrt(128)

    // ---- Q fragments: A-matrix 16x32 f16 layout, 4 chunks covering D=128 ----
    // lane row m = lh; K contiguous kb..kb+15 with kb = 16*hi
    v16h qf[4];
    {
        const float* qrow = Qg + (size_t)(bb * S_N + q0 + lh) * D_N;
#pragma unroll
        for (int c = 0; c < 4; ++c) {
            const float* qp = qrow + c * 32 + hi * 16;
            v4f a0 = *(const v4f*)(qp);
            v4f a1 = *(const v4f*)(qp + 4);
            v4f b0 = *(const v4f*)(qp + 8);
            v4f b1 = *(const v4f*)(qp + 12);
            v16h q;
#pragma unroll
            for (int i = 0; i < 4; ++i) {
                q[i]      = (_Float16)a0[i];
                q[4 + i]  = (_Float16)a1[i];
                q[8 + i]  = (_Float16)b0[i];
                q[12 + i] = (_Float16)b1[i];
            }
            qf[c] = q;
        }
    }

    // ---- accumulators: O (8 d-chunks of 16), online-softmax row state ----
    v8f zero = {};
    v8f acc[8];
#pragma unroll
    for (int j = 0; j < 8; ++j) acc[j] = zero;
    float rowmax[8], rowsum[8];
#pragma unroll
    for (int v = 0; v < 8; ++v) { rowmax[v] = -3.0e38f; rowsum[v] = 0.0f; }

    // cooperative staging assignment: 128 threads cover 32 rows x 4 col-groups
    const int sr = tid >> 2;          // key row 0..31
    const int sc = (tid & 3) * 32;    // 32-wide column group

    const int* mbase = Mg + (size_t)bb * S_N * S_N + (size_t)q0 * S_N;

    for (int t = 0; t < NT; ++t) {
        const int k0 = t * KT;

        // ---- stage K (row-major, packed b128 stores) and V (transposed) ----
        {
            const float* kp = Kg + (size_t)(bb * S_N + k0 + sr) * D_N + sc;
            const float* vp = Vg + (size_t)(bb * S_N + k0 + sr) * D_N + sc;
#pragma unroll
            for (int i = 0; i < 32; i += 8) {
                v4f k0v = *(const v4f*)(kp + i);
                v4f k1v = *(const v4f*)(kp + i + 4);
                v4f v0v = *(const v4f*)(vp + i);
                v4f v1v = *(const v4f*)(vp + i + 4);
                v8h kh;
#pragma unroll
                for (int u = 0; u < 4; ++u) {
                    kh[u]     = (_Float16)k0v[u];
                    kh[4 + u] = (_Float16)k1v[u];
                }
                *(v8h*)(&Kl[sr][sc + i]) = kh;   // one ds_store_b128
#pragma unroll
                for (int u = 0; u < 4; ++u) {    // scattered b16 (transpose)
                    Vt[sc + i + u][sr]     = (_Float16)v0v[u];
                    Vt[sc + i + 4 + u][sr] = (_Float16)v1v[u];
                }
            }
            if (t + 1 < NT) {
                __builtin_prefetch(kp + KT * D_N, 0, 0);   // global_prefetch_b8
                __builtin_prefetch(vp + KT * D_N, 0, 0);
            }
        }
        __syncthreads();

        // ---- scores: S = Q K^T (two 16-key halves, K-dim = D = 4 x 32) ----
        // hoist all B fragments first so DS latency overlaps the WMMA chain
        v16h kf0[4], kf1[4];
#pragma unroll
        for (int c = 0; c < 4; ++c) {
            kf0[c] = *(const v16h*)(&Kl[lh][c * 32 + hi * 16]);
            kf1[c] = *(const v16h*)(&Kl[16 + lh][c * 32 + hi * 16]);
        }
        v8f sA = zero, sB = zero;
#pragma unroll
        for (int c = 0; c < 4; ++c) {
            sA = __builtin_amdgcn_wmma_f32_16x16x32_f16(false, qf[c], false, kf0[c],
                                                        (short)0, sA, false, false);
            sB = __builtin_amdgcn_wmma_f32_16x16x32_f16(false, qf[c], false, kf1[c],
                                                        (short)0, sB, false, false);
        }

        // ---- mask + scale (C-layout: lane col = lh, rows = v + 8*hi) ----
        float s0[8], s1[8];
#pragma unroll
        for (int v = 0; v < 8; ++v) {
            const int m = v + hi * 8;
            const int* mrow = mbase + (size_t)m * S_N + k0;
            const int m0 = mrow[lh];
            const int m1 = mrow[16 + lh];
            s0[v] = (m0 == MASK_SENTINEL) ? -3.4e38f : sA[v] * scale;
            s1[v] = (m1 == MASK_SENTINEL) ? -3.4e38f : sB[v] * scale;
        }

        // ---- online softmax: row max/sum reduced across the 16-lane group ----
        float tmax[8];
#pragma unroll
        for (int v = 0; v < 8; ++v) tmax[v] = fmaxf(s0[v], s1[v]);
#pragma unroll
        for (int off = 1; off < 16; off <<= 1) {
#pragma unroll
            for (int v = 0; v < 8; ++v)
                tmax[v] = fmaxf(tmax[v], __shfl_xor(tmax[v], off, 32));
        }

        float p0[8], p1[8], corr[8], tsum[8];
#pragma unroll
        for (int v = 0; v < 8; ++v) {
            const float mn = fmaxf(rowmax[v], tmax[v]);
            corr[v]   = __expf(rowmax[v] - mn);
            rowmax[v] = mn;
            p0[v] = __expf(s0[v] - mn);
            p1[v] = __expf(s1[v] - mn);
            tsum[v] = p0[v] + p1[v];
        }
#pragma unroll
        for (int off = 1; off < 16; off <<= 1) {
#pragma unroll
            for (int v = 0; v < 8; ++v)
                tsum[v] += __shfl_xor(tsum[v], off, 32);
        }
#pragma unroll
        for (int v = 0; v < 8; ++v) rowsum[v] = rowsum[v] * corr[v] + tsum[v];
#pragma unroll
        for (int j = 0; j < 8; ++j) {
#pragma unroll
            for (int v = 0; v < 8; ++v) acc[j][v] *= corr[v];
        }

        // ---- reshape P (C-layout -> A-matrix 16x32) through LDS ----
        {
            _Float16* pw = &Pl[wave][0][0];
#pragma unroll
            for (int v = 0; v < 8; ++v) {
                const int m = v + hi * 8;
                pw[m * KT + lh]      = (_Float16)p0[v];
                pw[m * KT + 16 + lh] = (_Float16)p1[v];
            }
        }
        __syncthreads();

        v8h plo = *(const v8h*)(&Pl[wave][lh][hi * 16]);
        v8h phi = *(const v8h*)(&Pl[wave][lh][hi * 16 + 8]);
        v16h pf = __builtin_shufflevector(plo, phi,
                                          0, 1, 2, 3, 4, 5, 6, 7,
                                          8, 9, 10, 11, 12, 13, 14, 15);

        // ---- O += P V : hoist all 8 B-fragments, then the WMMA chain ----
        v16h vf[8];
#pragma unroll
        for (int j = 0; j < 8; ++j)
            vf[j] = *(const v16h*)(&Vt[j * 16 + lh][hi * 16]);
#pragma unroll
        for (int j = 0; j < 8; ++j) {
            acc[j] = __builtin_amdgcn_wmma_f32_16x16x32_f16(false, pf, false, vf[j],
                                                            (short)0, acc[j],
                                                            false, false);
        }
        __syncthreads();   // tiles fully consumed before next staging pass
    }

    // ---- epilogue: normalize by row sums and write O ----
#pragma unroll
    for (int v = 0; v < 8; ++v) {
        const float inv = 1.0f / rowsum[v];
        float* orow = Og + (size_t)(bb * S_N + q0 + hi * 8 + v) * D_N + lh;
#pragma unroll
        for (int j = 0; j < 8; ++j) orow[j * 16] = acc[j][v] * inv;
    }
}

extern "C" void kernel_launch(void* const* d_in, const int* in_sizes, int n_in,
                              void* d_out, int out_size, void* d_ws, size_t ws_size,
                              hipStream_t stream) {
    const float* Vg = (const float*)d_in[0];   // value
    const float* Kg = (const float*)d_in[1];   // key
    const float* Qg = (const float*)d_in[2];   // query
    const int*   Mg = (const int*)d_in[3];     // mask
    float* Og = (float*)d_out;

    dim3 grid(B_N * (S_N / (WAVES * 16)));     // 16 * 32 = 512 blocks
    dim3 block(WAVES * 32);                    // 4 waves of 32
    hipLaunchKernelGGL(fa_fwd_wmma, grid, block, 0, stream, Vg, Kg, Qg, Mg, Og);
}
